// ConvLSTM_4638564680394
// MI455X (gfx1250) — matrix operand order repaired and checked by reading, not compiled, not run on hardware
//
#include <hip/hip_runtime.h>
#include <hip/hip_bf16.h>
#include <hip/hip_fp16.h>

// ConvLSTM on gfx1250 (MI455X):
//  - implicit-GEMM 3x3 conv via v_wmma_f32_16x16x32_f16 (f16 in, f32 accum)
//  - h states kept f16 NHWC in global -> patch staged with
//    global_load_async_to_lds_b128 (ASYNCcnt), no VALU convert, no VGPR trip
//  - c states f32 NHWC (contiguous b128 gate-phase accesses)
//  - fused LSTM gate math per tile; ping-pong h buffers for the 3x3 halo

typedef __attribute__((ext_vector_type(16))) _Float16 v16h;
typedef __attribute__((ext_vector_type(8)))  _Float16 v8h;
typedef __attribute__((ext_vector_type(8)))  float    v8f;

#define BATCH 16
#define TSTEPS 16
#define HID 64
#define IMH 64
#define IMW 64
#define HW  (IMH * IMW)
#define MGATES 256   // 4*HID gate channels

__device__ __forceinline__ float sigf(float x) {
    return 1.0f / (1.0f + __expf(-x));
}

// Async global->LDS copy of 16 bytes (CDNA5 VGLOBAL async path, ASYNCcnt).
__device__ __forceinline__ void async_copy_b128(unsigned lds_off, const void* gptr) {
    asm volatile("global_load_async_to_lds_b128 %0, %1, off"
                 :: "v"(lds_off), "v"((unsigned long long)(uintptr_t)gptr)
                 : "memory");
}

__device__ __forceinline__ void wait_async0() {
#if __has_builtin(__builtin_amdgcn_s_wait_asynccnt)
    __builtin_amdgcn_s_wait_asynccnt(0);
#else
    asm volatile("s_wait_asynccnt 0" ::: "memory");
#endif
}

// ---------------------------------------------------------------------------
// Zero-init kernel (grid-stride)
// ---------------------------------------------------------------------------
__global__ void zero_kernel(float* __restrict__ p, size_t n) {
    size_t i = (size_t)blockIdx.x * blockDim.x + threadIdx.x;
    size_t stride = (size_t)gridDim.x * blockDim.x;
    for (; i < n; i += stride) p[i] = 0.0f;
}

// ---------------------------------------------------------------------------
// Weight repack: f32 OIHW -> f16 WMMA A-fragments.
// K ordering: k = r*Cpad + c (r = tap dy*3+dx, c = GEMM input channel).
// GEMM channel map: hswap=1 (layer0): c<64 -> w in-ch c+1 (h part),
//                   c==64 -> w in-ch 0 (x), c>64 -> zero pad.
//                   hswap=0 (layer1): identity, c<Cin.
// Fragment (kc, mtile): 32 lanes x 16 halves contiguous (32B/lane), matching
// the contiguous-k order this code uses for A and B (lane<16: M=lane, K half i -> i;
// lane>=16: M=lane-16, K +16).
// ---------------------------------------------------------------------------
__global__ void pack_weights(const float* __restrict__ w, _Float16* __restrict__ wpk,
                             int Cin, int Cpad, int nChunk, int hswap) {
    int gid = blockIdx.x * blockDim.x + threadIdx.x;
    int total = nChunk * 16 * 32;
    if (gid >= total) return;
    int lane = gid & 31;
    int frag = gid >> 5;              // = kc*16 + mtile
    int mtile = frag & 15;
    int kc = frag >> 4;
    int m = mtile * 16 + (lane & 15);
    int hi = (lane >= 16) ? 8 : 0;

    v16h tmp;
#pragma unroll
    for (int i = 0; i < 16; ++i) {
        int klocal = i + 2 * hi;
        int k = kc * 32 + klocal;
        int r = k / Cpad;             // tap 0..8
        int c = k % Cpad;             // GEMM channel
        int wc;
        if (hswap) wc = (c < 64) ? (c + 1) : ((c == 64) ? 0 : -1);
        else       wc = (c < Cin) ? c : -1;
        float v = (wc >= 0) ? w[(size_t)(m * Cin + wc) * 9 + r] : 0.0f;
        tmp[i] = (_Float16)v;
    }
    ((v16h*)wpk)[(size_t)frag * 32 + lane] = tmp;
}

// ---------------------------------------------------------------------------
// Fused conv(3x3,SAME) + LSTM cell step. Tile: 256 gate ch x 32 pixels.
// Grid (W/32, H, B), block 256 = 8 waves.
// GEMM input channels (CPAD padded):
//   HASX=true  (layer0, CPAD=96):  0..63 <- hA (f16 NHWC), 64 <- x (f32 NCHW),
//                                  65..95 zero
//   HASX=false (layer1, CPAD=128): 0..63 <- hA, 64..127 <- hB (both f16 NHWC)
// ---------------------------------------------------------------------------
template <int CPAD, bool HASX>
__global__ __launch_bounds__(256) void step_kernel(
    const _Float16* __restrict__ hA,
    const _Float16* __restrict__ hB,
    const float*    __restrict__ xt,      // + t*HW already; batch stride T*HW
    const _Float16* __restrict__ wpk,
    const float*    __restrict__ bias,
    float*          __restrict__ cstate,  // f32 NHWC, in-place
    _Float16*       __restrict__ h_out)   // f16 NHWC
{
    constexpr int CB = CPAD / 32;                        // K chunks per tap
    __shared__ __align__(32) _Float16 patch[3 * 34 * CPAD];  // [row*34+col][c]
    __shared__ float gatesS[MGATES * 32];                    // [m][n]

    const int tid = threadIdx.x;
    const int x0 = blockIdx.x * 32;
    const int y  = blockIdx.y;
    const int b  = blockIdx.z;

    // ---- Stage patch: async f16 chunks from h states (16B per chunk) ----
    {
        constexpr int NCH = HASX ? 8 : (CPAD / 8);       // 16B chunks per pos
        for (int idx = tid; idx < 102 * NCH; idx += 256) {
            int pos = idx / NCH;
            int c0 = (idx % NCH) * 8;                    // first channel
            int row = pos / 34, col = pos % 34;
            int gy = y + row - 1;
            int gx = x0 + col - 1;
            unsigned loff = (unsigned)(uintptr_t)&patch[pos * CPAD + c0];
            if (gy >= 0 && gy < IMH && gx >= 0 && gx < IMW) {
                size_t pix = ((size_t)b * HW + gy * IMW + gx) * HID;
                const _Float16* src = (c0 < HID) ? (hA + pix + c0)
                                                 : (hB + pix + (c0 - HID));
                async_copy_b128(loff, src);
            } else {
                *(v8h*)&patch[pos * CPAD + c0] = (v8h){};
            }
        }
        if constexpr (HASX) {
            // zero pad band: channels 65..CPAD-1 (always zero)
            for (int idx = tid; idx < 102 * (CPAD - 65); idx += 256) {
                int pos = idx / (CPAD - 65);
                int c = 65 + idx % (CPAD - 65);
                patch[pos * CPAD + c] = (_Float16)0.0f;
            }
            // x goes to channel 64 (f32 -> f16 convert, 102 scalars)
            for (int pos = tid; pos < 102; pos += 256) {
                int row = pos / 34, col = pos % 34;
                int gy = y + row - 1;
                int gx = x0 + col - 1;
                float v = 0.0f;
                if (gy >= 0 && gy < IMH && gx >= 0 && gx < IMW)
                    v = xt[(size_t)b * TSTEPS * HW + gy * IMW + gx];
                patch[pos * CPAD + 64] = (_Float16)v;
            }
        }
    }
    wait_async0();
    __syncthreads();

    // ---- Implicit GEMM: 256 x 32 x (9*CPAD) via WMMA ----
    const int wave = tid >> 5;
    const int lane = tid & 31;
    const int laneN = lane & 15;
    const int hiK = (lane >> 4) ? 16 : 0;
    const int m0 = wave * 2;
    const int m1 = m0 + 1;

    v8f acc0 = {}, acc1 = {}, acc2 = {}, acc3 = {};
    const v16h* wfrag = (const v16h*)wpk;

    for (int r = 0; r < 9; ++r) {
        const int dy = r / 3, dx = r % 3;
#pragma unroll
        for (int cb = 0; cb < CB; ++cb) {
            const int kc = r * CB + cb;
            v16h a0 = wfrag[((size_t)kc * 16 + m0) * 32 + lane];
            v16h a1 = wfrag[((size_t)kc * 16 + m1) * 32 + lane];
            __builtin_prefetch((const void*)&wfrag[((size_t)(kc + 1) * 16 + m0) * 32 + lane], 0, 3);
            const int col0 = laneN + dx;
            const int col1 = 16 + laneN + dx;
            const _Float16* p0 = &patch[(dy * 34 + col0) * CPAD + cb * 32 + hiK];
            const _Float16* p1 = &patch[(dy * 34 + col1) * CPAD + cb * 32 + hiK];
            v16h b0 = *(const v16h*)p0;
            v16h b1 = *(const v16h*)p1;
            acc0 = __builtin_amdgcn_wmma_f32_16x16x32_f16(false, a0, false, b0, (short)0, acc0, false, false);
            acc1 = __builtin_amdgcn_wmma_f32_16x16x32_f16(false, a0, false, b1, (short)0, acc1, false, false);
            acc2 = __builtin_amdgcn_wmma_f32_16x16x32_f16(false, a1, false, b0, (short)0, acc2, false, false);
            acc3 = __builtin_amdgcn_wmma_f32_16x16x32_f16(false, a1, false, b1, (short)0, acc3, false, false);
        }
    }

    // ---- Spill accumulators to LDS ([m][n]) ----
    // C layout: lanes 0-15 M=vgpr, lanes 16-31 M=vgpr+8; N = lane%16
    {
        const int mrow = (lane >> 4) ? 8 : 0;
#pragma unroll
        for (int j = 0; j < 8; ++j) {
            gatesS[(m0 * 16 + mrow + j) * 32 + (0 + laneN)]  = acc0[j];
            gatesS[(m0 * 16 + mrow + j) * 32 + (16 + laneN)] = acc1[j];
            gatesS[(m1 * 16 + mrow + j) * 32 + (0 + laneN)]  = acc2[j];
            gatesS[(m1 * 16 + mrow + j) * 32 + (16 + laneN)] = acc3[j];
        }
    }
    __syncthreads();

    // ---- LSTM gate math; i,f,o,g in M blocks 0/64/128/192 ----
    {
        const int n = tid & 31;
        const int hcb = (tid >> 5) * 8;
        const int gx = x0 + n;
        const size_t pix = ((size_t)b * HW + y * IMW + gx) * HID + hcb;
        v8f cold = *(const v8f*)(cstate + pix);
        v8f cnew;
        v8h hnew;
#pragma unroll
        for (int j = 0; j < 8; ++j) {
            int hc = hcb + j;
            float iv = gatesS[(hc)       * 32 + n] + bias[hc];
            float fv = gatesS[(64 + hc)  * 32 + n] + bias[64 + hc];
            float ov = gatesS[(128 + hc) * 32 + n] + bias[128 + hc];
            float gv = gatesS[(192 + hc) * 32 + n] + bias[192 + hc];
            float cn = sigf(fv) * cold[j] + sigf(iv) * tanhf(gv);
            cnew[j] = cn;
            hnew[j] = (_Float16)(sigf(ov) * tanhf(cn));
        }
        *(v8f*)(cstate + pix) = cnew;
        *(v8h*)(h_out + pix)  = hnew;
    }
}

// ---------------------------------------------------------------------------
// Head: 1x1 conv (HID->1) + ReLU over f16 NHWC h1
// ---------------------------------------------------------------------------
__global__ __launch_bounds__(256) void head_kernel(const _Float16* __restrict__ h1,
                                                   const float* __restrict__ wh,
                                                   const float* __restrict__ bh,
                                                   float* __restrict__ out) {
    __shared__ float wsm[HID];
    if (threadIdx.x < HID) wsm[threadIdx.x] = wh[threadIdx.x];
    __syncthreads();
    int gid = blockIdx.x * 256 + threadIdx.x;          // 0 .. B*HW-1
    const _Float16* base = h1 + (size_t)gid * HID;     // contiguous 128B
    float acc = bh[0];
#pragma unroll
    for (int c = 0; c < HID; ++c) acc = fmaf(wsm[c], (float)base[c], acc);
    out[gid] = fmaxf(acc, 0.0f);
}

// ---------------------------------------------------------------------------
// Launcher
// ---------------------------------------------------------------------------
extern "C" void kernel_launch(void* const* d_in, const int* in_sizes, int n_in,
                              void* d_out, int out_size, void* d_ws, size_t ws_size,
                              hipStream_t stream) {
    const float* x  = (const float*)d_in[0];
    const float* w0 = (const float*)d_in[1];
    const float* b0 = (const float*)d_in[2];
    const float* w1 = (const float*)d_in[3];
    const float* b1 = (const float*)d_in[4];
    const float* wh = (const float*)d_in[5];
    const float* bh = (const float*)d_in[6];
    float* out = (float*)d_out;

    const size_t S = (size_t)BATCH * HW * HID;         // elements per state
    float* ws = (float*)d_ws;
    // Layout: [c0 (f32), c1 (f32), h0a|h1a (f16, zeroed), h0b|h1b (f16), wpk]
    float* c0 = ws;
    float* c1 = ws + S;
    _Float16* h0a = (_Float16*)(ws + 2 * S);
    _Float16* h1a = h0a + S;
    _Float16* h0b = h0a + 2 * S;
    _Float16* h1b = h0a + 3 * S;
    _Float16* wpk0 = h0a + 4 * S;
    const int nChunk0 = 9 * (96 / 32);                 // 27
    const int nChunk1 = 9 * (128 / 32);                // 36
    _Float16* wpk1 = wpk0 + (size_t)nChunk0 * 16 * 32 * 16;

    // 1) zero c0, c1, h0a, h1a (contiguous: 2S floats + 2S halves = 3S floats)
    zero_kernel<<<4096, 256, 0, stream>>>(ws, 3 * S);

    // 2) pack weights into WMMA A fragments
    {
        int t0 = nChunk0 * 16 * 32;
        int t1 = nChunk1 * 16 * 32;
        pack_weights<<<(t0 + 255) / 256, 256, 0, stream>>>(w0, wpk0, 65, 96, nChunk0, 1);
        pack_weights<<<(t1 + 255) / 256, 256, 0, stream>>>(w1, wpk1, 128, 128, nChunk1, 0);
    }

    // 3) recurrence (ping-pong h buffers; c updated in place)
    dim3 grid(IMW / 32, IMH, BATCH);
    _Float16* h0r = h0a; _Float16* h0w = h0b;
    _Float16* h1r = h1a; _Float16* h1w = h1b;
    for (int t = 0; t < TSTEPS; ++t) {
        // layer 0: channels [h0r(64), x(1), pad] -> writes c0, h0w
        step_kernel<96, true><<<grid, 256, 0, stream>>>(
            h0r, (const _Float16*)nullptr, x + (size_t)t * HW, wpk0, b0, c0, h0w);
        // layer 1: channels [h0w(64), h1r(64)] -> writes c1, h1w
        step_kernel<128, false><<<grid, 256, 0, stream>>>(
            h0w, h1r, (const float*)nullptr, wpk1, b1, c1, h1w);
        _Float16* tmp;
        tmp = h0r; h0r = h0w; h0w = tmp;
        tmp = h1r; h1r = h1w; h1w = tmp;
    }

    // 4) head: 1x1 conv + ReLU over final h1 (in h1r after last swap)
    head_kernel<<<(BATCH * HW) / 256, 256, 0, stream>>>(h1r, wh, bh, out);
}
